// RetriContrastLoss_3908420239914
// MI455X (gfx1250) — matrix operand order repaired and checked by reading, not compiled, not run on hardware
//
#include <hip/hip_runtime.h>
#include <hip/hip_bf16.h>

// CDNA5 / gfx1250: wave32, WMMA f32_16x16x32_f16 + GLOBAL_LOAD_ASYNC_TO_LDS path.
typedef __attribute__((ext_vector_type(16))) _Float16 v16h;
typedef __attribute__((ext_vector_type(8)))  float    v8f;
typedef __attribute__((ext_vector_type(4)))  _Float16 v4h;

#define TILE_BM 256        // A rows per block
#define TILE_BN 128        // B rows (output cols) per block
#define TILE_K  32
#define LDS_STRIDE 40      // 32 data halves + 8 pad halves -> conflict-free frag loads
#define MARGIN 0.5f
#define SHIFT  3.75f       // max circle-transformed score: (1+m)*(1 + (1-m)) = 2.5*1.5

union FragU { v16h v; uint4 u[2]; };

__device__ __forceinline__ float wave_reduce_add(float x) {
#pragma unroll
  for (int off = 16; off > 0; off >>= 1) x += __shfl_down(x, off, 32);
  return x;
}

// Async DMA: global -> LDS, 16B per lane, tracked by ASYNCcnt (no VGPR staging).
// VDST takes the wave-relative LDS byte address = low 32 bits of the generic pointer.
__device__ __forceinline__ void async_copy_b128(void* lds_dst, const void* gsrc) {
  const uint32_t lds_off = (uint32_t)(uintptr_t)lds_dst;
  asm volatile("global_load_async_to_lds_b128 %0, %1, off"
               :: "v"(lds_off), "v"(gsrc) : "memory");
}
__device__ __forceinline__ void wait_asynccnt0() {
  asm volatile("s_wait_asynccnt 0x0" ::: "memory");
}

__global__ void init_sums_kernel(float* sums) {
  if (threadIdx.x < 2) sums[threadIdx.x] = 0.0f;
}

// One block per row; blockIdx.y selects q (0) or n (1). Assumes D == 4*blockDim.x.
__global__ __launch_bounds__(256) void normalize_to_f16_kernel(
    const float* __restrict__ q, const float* __restrict__ n,
    _Float16* __restrict__ qh, _Float16* __restrict__ nh, int D) {
  const float* src = blockIdx.y ? n : q;
  _Float16*    dst = blockIdx.y ? nh : qh;
  const int row = blockIdx.x;
  const int tid = threadIdx.x;
  const float4* p = (const float4*)(src + (size_t)row * D);
  float4 v = p[tid];
  float ss = v.x * v.x + v.y * v.y + v.z * v.z + v.w * v.w;
  ss = wave_reduce_add(ss);
  __shared__ float red[8];
  __shared__ float s_inv;
  if ((tid & 31) == 0) red[tid >> 5] = ss;
  __syncthreads();
  if (tid == 0) {
    float t = 0.0f;
#pragma unroll
    for (int w = 0; w < 8; ++w) t += red[w];
    s_inv = 1.0f / fmaxf(sqrtf(t), 1e-8f);
  }
  __syncthreads();
  const float inv = s_inv;
  v4h h;
  h[0] = (_Float16)(v.x * inv);
  h[1] = (_Float16)(v.y * inv);
  h[2] = (_Float16)(v.z * inv);
  h[3] = (_Float16)(v.w * inv);
  *(v4h*)(dst + (size_t)row * D + (size_t)tid * 4) = h;
}

// Per-row cosine(q,p) -> circle transform -> exp(s - SHIFT) accumulated atomically.
__global__ __launch_bounds__(256) void pos_lse_kernel(
    const float* __restrict__ q, const float* __restrict__ p,
    float* __restrict__ sum_pos, int D) {
  const int row = blockIdx.x, tid = threadIdx.x;
  const float4* qp = (const float4*)(q + (size_t)row * D);
  const float4* pp = (const float4*)(p + (size_t)row * D);
  float4 a = qp[tid], b = pp[tid];
  float d  = a.x * b.x + a.y * b.y + a.z * b.z + a.w * b.w;
  float qq = a.x * a.x + a.y * a.y + a.z * a.z + a.w * a.w;
  float bb = b.x * b.x + b.y * b.y + b.z * b.z + b.w * b.w;
  d = wave_reduce_add(d);
  qq = wave_reduce_add(qq);
  bb = wave_reduce_add(bb);
  __shared__ float rd[8], rq[8], rp[8];
  if ((tid & 31) == 0) { rd[tid >> 5] = d; rq[tid >> 5] = qq; rp[tid >> 5] = bb; }
  __syncthreads();
  if (tid == 0) {
    float D0 = 0.f, Q0 = 0.f, P0 = 0.f;
#pragma unroll
    for (int w = 0; w < 8; ++w) { D0 += rd[w]; Q0 += rq[w]; P0 += rp[w]; }
    const float s  = D0 / (fmaxf(sqrtf(Q0), 1e-8f) * fmaxf(sqrtf(P0), 1e-8f));
    const float ap = fmaxf(-s + (1.0f + MARGIN), 0.0f);
    const float sp = -ap * (s - (1.0f - MARGIN));
    atomicAdd(sum_pos, __expf(sp - SHIFT));
  }
}

// 256x128 output tile per block; 8 waves (4x2), each owning a 64x64 patch (4x4 WMMA tiles).
// Double-buffered LDS fed by GLOBAL_LOAD_ASYNC_TO_LDS_B128 (ASYNCcnt), fused circle
// transform + fixed-shift streaming exp-sum so sim_neg never touches memory.
__global__ __launch_bounds__(256) void wmma_circle_lse_kernel(
    const _Float16* __restrict__ Qh, const _Float16* __restrict__ Nh,
    float* __restrict__ sum_neg, int D) {
  __shared__ _Float16 As[2][TILE_BM * LDS_STRIDE];
  __shared__ _Float16 Bs[2][TILE_BN * LDS_STRIDE];
  __shared__ float red[8];

  const int tid    = threadIdx.x;
  const int wid    = tid >> 5;
  const int lane   = tid & 31;
  const int wave_m = wid >> 1;       // 0..3 -> 64-row strips
  const int wave_n = wid & 1;        // 0..1 -> 64-col strips
  const int r      = lane & 15;
  const int hi     = lane >> 4;      // half-wave select for K layout

  const _Float16* Abase = Qh + (size_t)(blockIdx.y * TILE_BM) * D;
  const _Float16* Bbase = Nh + (size_t)(blockIdx.x * TILE_BN) * D;

  // Cooperative async tile fill: 16B chunks, 4 per row; A: 1024 chunks, B: 512 chunks.
  const int NK = D / TILE_K;
  v8f acc[4][4] = {};

  {
    // prologue: buffer 0, k = 0
#pragma unroll
    for (int i = 0; i < 4; ++i) {
      const int c = tid + 256 * i, row = c >> 2, col = c & 3;
      async_copy_b128(&As[0][row * LDS_STRIDE + col * 8],
                      (const uint4*)(Abase + (size_t)row * D) + col);
    }
#pragma unroll
    for (int i = 0; i < 2; ++i) {
      const int c = tid + 256 * i, row = c >> 2, col = c & 3;
      async_copy_b128(&Bs[0][row * LDS_STRIDE + col * 8],
                      (const uint4*)(Bbase + (size_t)row * D) + col);
    }
  }

  int cur = 0;
  for (int k = 0; k < NK; ++k) {
    wait_asynccnt0();     // my buf[cur] writes are in LDS
    __syncthreads();      // everyone's buf[cur] writes are in LDS; buf[cur^1] readers done

    if (k + 1 < NK) {
      const int ku4 = (k + 1) * (TILE_K / 8);
      const int nxt = cur ^ 1;
#pragma unroll
      for (int i = 0; i < 4; ++i) {
        const int c = tid + 256 * i, row = c >> 2, col = c & 3;
        async_copy_b128(&As[nxt][row * LDS_STRIDE + col * 8],
                        (const uint4*)(Abase + (size_t)row * D) + ku4 + col);
      }
#pragma unroll
      for (int i = 0; i < 2; ++i) {
        const int c = tid + 256 * i, row = c >> 2, col = c & 3;
        async_copy_b128(&Bs[nxt][row * LDS_STRIDE + col * 8],
                        (const uint4*)(Bbase + (size_t)row * D) + ku4 + col);
      }
      if (k + 2 < NK) {   // warm GL2 one tile further ahead
        __builtin_prefetch((const uint4*)(Abase + (size_t)(tid >> 2) * D) +
                               (k + 2) * (TILE_K / 8) + (tid & 3), 0, 1);
      }
    }

    // A fragment (16x32 f16): lanes 0-15 hold M=r, K {0..7,16..23}; lanes 16-31 K {8..15,24..31}.
    FragU af[4], bf[4];
#pragma unroll
    for (int mt = 0; mt < 4; ++mt) {
      const uint4* pa = (const uint4*)&As[cur][(wave_m * 64 + mt * 16 + r) * LDS_STRIDE];
      af[mt].u[0] = pa[hi];
      af[mt].u[1] = pa[hi + 2];
    }
    // B fragment (32x16 f16): lanes 0-15 hold N=r, K {0..7,16..23}; lanes 16-31 K {8..15,24..31}.
#pragma unroll
    for (int nt = 0; nt < 4; ++nt) {
      const uint4* pb = (const uint4*)&Bs[cur][(wave_n * 64 + nt * 16 + r) * LDS_STRIDE];
      bf[nt].u[0] = pb[hi];
      bf[nt].u[1] = pb[hi + 2];
    }

#pragma unroll
    for (int mt = 0; mt < 4; ++mt)
#pragma unroll
      for (int nt = 0; nt < 4; ++nt)
        acc[mt][nt] = __builtin_amdgcn_wmma_f32_16x16x32_f16(
            false, af[mt].v, false, bf[nt].v, (short)0, acc[mt][nt], false, false);

    cur ^= 1;
  }

  // Fused epilogue: circle transform + exp with fixed shift (scores <= 3.75).
  float part = 0.0f;
#pragma unroll
  for (int mt = 0; mt < 4; ++mt)
#pragma unroll
    for (int nt = 0; nt < 4; ++nt)
#pragma unroll
      for (int e = 0; e < 8; ++e) {
        const float s  = acc[mt][nt][e];
        const float an = fmaxf(s + (1.0f + MARGIN), 0.0f);
        const float sn = an * (s + (1.0f - MARGIN));
        part += __expf(sn - SHIFT);
      }
  part = wave_reduce_add(part);
  if (lane == 0) red[wid] = part;
  __syncthreads();
  if (tid == 0) {
    float t = 0.0f;
#pragma unroll
    for (int w = 0; w < 8; ++w) t += red[w];
    atomicAdd(sum_neg, t);
  }
}

__global__ void finalize_kernel(const float* sums, float* out) {
  if (threadIdx.x == 0 && blockIdx.x == 0) {
    const float lse_neg = logf(sums[0]) + SHIFT;
    const float lse_pos = logf(sums[1]) + SHIFT;
    const float z = lse_neg + lse_pos;
    out[0] = (z > 0.0f) ? (z + log1pf(expf(-z))) : log1pf(expf(z));
  }
}

extern "C" void kernel_launch(void* const* d_in, const int* in_sizes, int n_in,
                              void* d_out, int out_size, void* d_ws, size_t ws_size,
                              hipStream_t stream) {
  const float* q = (const float*)d_in[0];
  const float* p = (const float*)d_in[1];
  const float* n = (const float*)d_in[2];
  // d_in[3] (text_neg_index) is unused by the reference.
  const int N = in_sizes[3];           // 8192 rows
  const int D = in_sizes[0] / N;       // 1024
  float* out = (float*)d_out;

  char* ws = (char*)d_ws;
  float* sums = (float*)ws;            // [0] = sum_neg, [1] = sum_pos
  _Float16* qh = (_Float16*)(ws + 256);
  _Float16* nh = qh + (size_t)N * D;
  const size_t needed = 256 + 2ull * (size_t)N * D * sizeof(_Float16);
  if (ws_size < needed) return;        // scratch too small; nothing safe to do

  init_sums_kernel<<<1, 32, 0, stream>>>(sums);
  normalize_to_f16_kernel<<<dim3(N, 2), 256, 0, stream>>>(q, n, qh, nh, D);
  pos_lse_kernel<<<N, 256, 0, stream>>>(q, p, sums + 1, D);
  wmma_circle_lse_kernel<<<dim3(N / TILE_BN, N / TILE_BM), 256, 0, stream>>>(qh, nh, sums, D);
  finalize_kernel<<<1, 32, 0, stream>>>(sums, out);
}
